// CausalAttention_72249939853363
// MI455X (gfx1250) — matrix operand order repaired and checked by reading, not compiled, not run on hardware
//
#include <hip/hip_runtime.h>
#include <hip/hip_bf16.h>

// ---------------------------------------------------------------------------
// Causal attention, decomposed:
//   1) convert X, [Wq;Wk;Wv] fp32 -> bf16
//   2) QKV = Xb @ Wb^T              (WMMA bf16, f32 accum, bf16 out)
//   3) Sc_b = Q_b @ K_b^T           (WMMA, causal upper-tile skip, fp32 out)
//   4) P = softmax(Sc * scale) rows (fp32 in, bf16 out, causal)
//   5) out_b = P_b @ V_b            (WMMA, causal K-limit, fp32 out)
// GEMM staging uses the Tensor Data Mover (tensor_load_to_lds) with
// double-buffered LDS so the DMA of tile i+1 overlaps WMMA on tile i.
// ---------------------------------------------------------------------------

typedef __attribute__((ext_vector_type(16))) __bf16        v16bf;
typedef __attribute__((ext_vector_type(8)))  float         v8f;
typedef __attribute__((ext_vector_type(8)))  unsigned short v8u;
typedef __attribute__((ext_vector_type(4)))  unsigned int  v4u;
typedef __attribute__((ext_vector_type(8)))  int           v8i;
typedef __attribute__((ext_vector_type(4)))  int           v4i;

#ifndef __has_builtin
#define __has_builtin(x) 0
#endif
#if defined(__gfx1250__) && __has_builtin(__builtin_amdgcn_tensor_load_to_lds) && \
    __has_builtin(__builtin_amdgcn_s_wait_tensorcnt)
#define USE_TDM 1
#else
#define USE_TDM 0
#endif

union FragAB { v16bf v; v8u h[2]; };

__device__ __forceinline__ unsigned short f2bf(float f) {
  unsigned int u = __builtin_bit_cast(unsigned int, f);
  unsigned int r = u + 0x7fffu + ((u >> 16) & 1u);   // round to nearest even
  return (unsigned short)(r >> 16);
}

// ----------------------------- convert --------------------------------------
__global__ void f32_to_bf16_kern(const float* __restrict__ src,
                                 unsigned short* __restrict__ dst, long n) {
  long i = (long)blockIdx.x * blockDim.x + threadIdx.x;
  long stride = (long)gridDim.x * blockDim.x;
  for (; i < n; i += stride) dst[i] = f2bf(src[i]);
}

// ----------------------------- WMMA GEMM ------------------------------------
constexpr int TM = 128, TN = 128, TK = 32;
constexpr int APITCH = 40, BPITCH = 40;   // bf16 elems; 80B rows keep 16B align
// 64B of payload per row + 16B pad == pitch 40 elems; TDM pads for us.

#if USE_TDM
// Issue one TDM load of a [128 rows x 32 cols] bf16 tile (row stride = lda
// elements) into LDS at `ldsDst`, padding each 64B row with 16B to match
// the 40-element LDS pitch.  D# packed per CDNA5 ISA ch.8.
__device__ __forceinline__ void tdm_load_tile(unsigned short* ldsDst,
                                              const unsigned short* gsrc,
                                              unsigned lda) {
  unsigned lds = (unsigned)(unsigned long long)(void*)ldsDst; // low32 = LDS addr
  unsigned long long ga = (unsigned long long)gsrc;
  v4u g0;
  g0[0] = 1u;                                        // count=1, user descriptor
  g0[1] = lds;                                       // lds_addr
  g0[2] = (unsigned)ga;                              // global_addr[31:0]
  g0[3] = (unsigned)((ga >> 32) & 0x1FFFFFFu) | (2u << 30);  // addr[56:32],type=2
  const unsigned td0 = 1u << 24, td1 = 1u << 24;     // generous dims (no OOB)
  const unsigned tile0 = 32u, tile1 = 128u;
  const unsigned long long s0 = (unsigned long long)lda;     // dim0 stride
  v8i g1;
  // data_size=2B (1<<16), pad_enable (1<<20), pad_interval=16DW (3<<22),
  // pad_amount=4DW (3<<25)
  g1[0] = (int)((1u << 16) | (1u << 20) | (3u << 22) | (3u << 25));
  g1[1] = (int)((td0 & 0xFFFFu) << 16);                      // tensor_dim0 lo16
  g1[2] = (int)((td0 >> 16) | ((td1 & 0xFFFFu) << 16));      // dim0 hi, dim1 lo
  g1[3] = (int)((td1 >> 16) | (tile0 << 16));                // dim1 hi, tile_dim0
  g1[4] = (int)(tile1);                                      // tile_dim1, tile_dim2=0
  g1[5] = (int)(unsigned)s0;                                 // stride0[31:0]
  g1[6] = (int)((unsigned)(s0 >> 32) & 0xFFFFu);             // stride0[47:32]
  g1[7] = 0;
  v4i z4 = {0, 0, 0, 0};
  v8i z8 = {0, 0, 0, 0, 0, 0, 0, 0};
  __builtin_amdgcn_tensor_load_to_lds(g0, g1, z4, z4, z8, 0);
}
#endif

// C[M,N] = A[M,K] * B, where B is stored [N,K] if BT else [K,N].
// Block tile 128x128, K-step 32, 8 wave32s (wave grid 4(m) x 2(n)),
// each wave: 2x4 fragments of 16x16 (v_wmma_f32_16x16x32_bf16).
template <bool BT, bool OUT_BF16, bool CSKIP, bool CKLIM>
__global__ __launch_bounds__(256)
void gemm_wmma_bf16(const unsigned short* __restrict__ A, int lda,
                    const unsigned short* __restrict__ B, int ldb,
                    void* __restrict__ C, int ldc, int K,
                    long batchA, long batchB, long batchC) {
  const int bm = blockIdx.y, bn = blockIdx.x, bz = blockIdx.z;
  if (CSKIP && bn > bm) return;   // tile fully above causal diagonal: never read

  const unsigned short* Ab = A + (long)bz * batchA + (long)(bm * TM) * lda;
  const unsigned short* Bb = B + (long)bz * batchB;

  __shared__ __align__(16) unsigned short As[2 * TM * APITCH];
  __shared__ __align__(16) unsigned short Bs[2 * TN * BPITCH];   // [n][k]

  const int tid  = threadIdx.x;
  const int wid  = tid >> 5, lane = tid & 31;
  const int wm   = wid >> 1, wn   = wid & 1;     // 4x2 wave grid
  const int lrow = lane & 15, lgrp = lane >> 4;

  v8f acc[2][4] = {};

  int kend = K;
  if (CKLIM) { int ke = (bm + 1) * TM; kend = ke < K ? ke : K; }
  const int nIter = kend / TK;

  const int srow = tid >> 1, scol = (tid & 1) * 16;   // manual staging indices

  auto stage = [&](int it) {
    const int k0  = it * TK;
    unsigned short* Asb = As + (it & 1) * (TM * APITCH);
    unsigned short* Bsb = Bs + (it & 1) * (TN * BPITCH);
#if USE_TDM
    if (tid < 32) {                       // one wave drives the DMA engine
      tdm_load_tile(Asb, Ab + k0, (unsigned)lda);
      if (BT) tdm_load_tile(Bsb, Bb + (long)(bn * TN) * ldb + k0, (unsigned)ldb);
    }
#else
    { // manual A stage: 32B per thread
      const v8u* src = (const v8u*)(Ab + (long)srow * lda + k0 + scol);
      v8u a0 = src[0], a1 = src[1];
      v8u* dst = (v8u*)(Asb + srow * APITCH + scol);
      dst[0] = a0; dst[1] = a1;
    }
    if (BT) {
      const v8u* src = (const v8u*)(Bb + (long)(bn * TN + srow) * ldb + k0 + scol);
      v8u b0 = src[0], b1 = src[1];
      v8u* dst = (v8u*)(Bsb + srow * BPITCH + scol);
      dst[0] = b0; dst[1] = b1;
    }
#endif
    if (!BT) {  // B stored [K,N]: transpose into [n][k] (TDM cannot transpose)
      const int bk = tid >> 3, bn0 = (tid & 7) * 16;
      const v8u* src = (const v8u*)(Bb + (long)(k0 + bk) * ldb + bn * TN + bn0);
      v8u b0 = src[0], b1 = src[1];
      unsigned short tmp[16];
      *(v8u*)tmp = b0; *(v8u*)(tmp + 8) = b1;
#pragma unroll
      for (int i = 0; i < 16; ++i) Bsb[(bn0 + i) * BPITCH + bk] = tmp[i];
    }
  };

  stage(0);
  for (int it = 0; it < nIter; ++it) {
    __syncthreads();                       // prior reads of next buffer done
    if (it + 1 < nIter) stage(it + 1);     // prefetch one tile ahead
#if USE_TDM
    if (tid < 32) {                        // tile `it` landed in LDS?
      if (it + 1 < nIter) {
        if (BT) __builtin_amdgcn_s_wait_tensorcnt(2);
        else    __builtin_amdgcn_s_wait_tensorcnt(1);
      } else {
        __builtin_amdgcn_s_wait_tensorcnt(0);
      }
    }
#endif
    __syncthreads();                       // buffer (it&1) visible to all waves

    const unsigned short* Asb = As + (it & 1) * (TM * APITCH);
    const unsigned short* Bsb = Bs + (it & 1) * (TN * BPITCH);

    // A fragment (16x32): lane m=lrow; grp0 K {0..7,16..23}, grp1 {8..15,24..31}
    FragAB afr[2], bfr[4];
#pragma unroll
    for (int i = 0; i < 2; ++i) {
      const v8u* p = (const v8u*)(Asb + (wm * 32 + i * 16 + lrow) * APITCH);
      afr[i].h[0] = p[lgrp];
      afr[i].h[1] = p[2 + lgrp];
    }
    // B fragment (32x16): lane n=lrow; grp0 K {0..7,16..23}, grp1 {8..15,24..31}
#pragma unroll
    for (int j = 0; j < 4; ++j) {
      const v8u* p = (const v8u*)(Bsb + (wn * 64 + j * 16 + lrow) * BPITCH);
      bfr[j].h[0] = p[lgrp];
      bfr[j].h[1] = p[2 + lgrp];
    }
#pragma unroll
    for (int i = 0; i < 2; ++i)
#pragma unroll
      for (int j = 0; j < 4; ++j)
        acc[i][j] = __builtin_amdgcn_wmma_f32_16x16x32_bf16(
            false, afr[i].v, false, bfr[j].v, (short)0, acc[i][j], false, false);
  }

  // Writeback: C fragment element e -> row M = 8*lgrp + e, col N = lrow
#pragma unroll
  for (int i = 0; i < 2; ++i) {
    const int mbase = bm * TM + wm * 32 + i * 16 + lgrp * 8;
#pragma unroll
    for (int j = 0; j < 4; ++j) {
      const int col = bn * TN + wn * 64 + j * 16 + lrow;
#pragma unroll
      for (int e = 0; e < 8; ++e) {
        const long idx = (long)bz * batchC + (long)(mbase + e) * ldc + col;
        if constexpr (OUT_BF16)
          ((unsigned short*)C)[idx] = f2bf(acc[i][j][e]);
        else
          ((float*)C)[idx] = acc[i][j][e];
      }
    }
  }
}

// ----------------------------- causal softmax -------------------------------
// One 256-thread block per row r = b*4096 + q. Reads raw fp32 scores cols 0..q,
// writes bf16 probs; zeros cols q+1..(q|127) so PV GEMM can K-limit per tile.
__global__ __launch_bounds__(256)
void softmax_causal(const float* __restrict__ Sc, unsigned short* __restrict__ P,
                    float scale) {
  const long r = blockIdx.x;
  const float* srow = Sc + r * 4096;
  unsigned short* prow = P + r * 4096;
  const int q = (int)(r & 4095);
  const int n = q + 1;
  const int tid = threadIdx.x;

  float cache[16];
  int cnt = 0;
  float mx = -3.402823466e38f;
  for (int c = tid; c < n; c += 256) {
    float v = srow[c];
    cache[cnt++] = v;
    mx = v > mx ? v : mx;
  }
  __shared__ float red[256];
  red[tid] = mx;
  __syncthreads();
  for (int s = 128; s > 0; s >>= 1) {
    if (tid < s) { float o = red[tid + s]; if (o > red[tid]) red[tid] = o; }
    __syncthreads();
  }
  mx = red[0];
  __syncthreads();

  float sum = 0.f;
  for (int i = 0; i < cnt; ++i) {
    float e = __expf((cache[i] - mx) * scale);
    cache[i] = e;
    sum += e;
  }
  red[tid] = sum;
  __syncthreads();
  for (int s = 128; s > 0; s >>= 1) {
    if (tid < s) red[tid] += red[tid + s];
    __syncthreads();
  }
  const float inv = 1.0f / red[0];

  int i = 0;
  for (int c = tid; c < n; c += 256) prow[c] = f2bf(cache[i++] * inv);
  const int zend = (q | 127) + 1;
  for (int c = n + tid; c < zend; c += 256) prow[c] = 0;
}

// ----------------------------- launcher -------------------------------------
extern "C" void kernel_launch(void* const* d_in, const int* in_sizes, int n_in,
                              void* d_out, int out_size, void* d_ws, size_t ws_size,
                              hipStream_t stream) {
  (void)in_sizes; (void)n_in; (void)out_size; (void)ws_size;
  constexpr long B = 4, S = 4096, D = 1024, BS = B * S;   // 16384

  const float* X  = (const float*)d_in[0];
  const float* Wq = (const float*)d_in[1];
  const float* Wk = (const float*)d_in[2];
  const float* Wv = (const float*)d_in[3];
  float* out = (float*)d_out;

  // workspace layout (all 16B aligned)
  unsigned short* Xb  = (unsigned short*)d_ws;             // [16384,1024] bf16
  unsigned short* Wb  = Xb + BS * D;                       // [3072,1024]  bf16
  unsigned short* QKV = Wb + 3 * D * D;                    // [16384,3072] bf16
  float*          Sc  = (float*)(QKV + BS * 3 * D);        // [4,4096,4096] f32
  unsigned short* P   = (unsigned short*)(Sc + B * S * S); // [4,4096,4096] bf16

  // 1) convert inputs to bf16
  f32_to_bf16_kern<<<2048, 256, 0, stream>>>(X,  Xb, BS * D);
  f32_to_bf16_kern<<<512,  256, 0, stream>>>(Wq, Wb,             D * D);
  f32_to_bf16_kern<<<512,  256, 0, stream>>>(Wk, Wb + D * D,     D * D);
  f32_to_bf16_kern<<<512,  256, 0, stream>>>(Wv, Wb + 2 * D * D, D * D);

  // 2) QKV projection: [16384,1024] x [3072,1024]^T -> bf16 [16384,3072]
  {
    dim3 grid(3072 / TN, BS / TM, 1);
    gemm_wmma_bf16<true, true, false, false><<<grid, 256, 0, stream>>>(
        Xb, (int)D, Wb, (int)D, QKV, 3072, (int)D, 0, 0, 0);
  }
  // 3) scores: per batch, Q_b [4096,1024] x K_b^T -> fp32, skip upper tiles
  {
    dim3 grid(S / TN, S / TM, B);
    gemm_wmma_bf16<true, false, true, false><<<grid, 256, 0, stream>>>(
        QKV + 0, 3072, QKV + 1024, 3072, Sc, (int)S, (int)D,
        S * 3072, S * 3072, S * S);
  }
  // 4) causal softmax rows (scale folded into exp)
  softmax_causal<<<(int)BS, 256, 0, stream>>>(Sc, P, 0.03125f);

  // 5) context: per batch, P_b [4096,4096] x V_b [4096,1024] -> fp32 out
  {
    dim3 grid(D / TN, S / TM, B);
    gemm_wmma_bf16<false, false, false, true><<<grid, 256, 0, stream>>>(
        P, (int)S, QKV + 2048, 3072, out, (int)D, (int)S,
        S * S, S * 3072, S * D);
  }
}
